// Method_38448547233834
// MI455X (gfx1250) — matrix operand order repaired and checked by reading, not compiled, not run on hardware
//
#include <hip/hip_runtime.h>
#include <hip/hip_bf16.h>

typedef _Float16 half_t;
typedef __attribute__((ext_vector_type(8)))  _Float16 v8h;
typedef __attribute__((ext_vector_type(16))) _Float16 v16h;
typedef __attribute__((ext_vector_type(8)))  float    v8f;

#define NC     3840
#define FS2    25
#define NWAY   5
#define NQ     75
#define NLEV   4
#define NTILE  120          // NC / 32 channel blocks
#define TILE_E 1024         // 32 pos x 32 k elements per tile
#define TILE_V 128          // v8h segments per tile (tile row = 4 segments)

// ---------------------------------------------------------------------------
// Kernel 1: channel-centering + per-level L2 norms, writing a WMMA-native
// tiled f16 layout: dst[((img*NTILE + c/32)*32 + pos)*32 + (c%32)].
// Pad rows pos=25..31 are pre-zeroed by hipMemsetAsync.
// One block per (image, spatial position).
// ---------------------------------------------------------------------------
__global__ __launch_bounds__(256)
void prep_kernel(const float* __restrict__ src, half_t* __restrict__ dstc,
                 float* __restrict__ norms)
{
    __shared__ float red[256];
    __shared__ float redl[4][256];
    const int b   = blockIdx.x;
    const int img = b / FS2;
    const int s   = b % FS2;
    const int t   = threadIdx.x;
    const float* col = src + (size_t)img * NC * FS2 + s;

    // channel mean
    float sum = 0.f;
    for (int c = t; c < NC; c += 256) sum += col[(size_t)c * FS2];
    red[t] = sum;
    __syncthreads();
    for (int o = 128; o > 0; o >>= 1) {
        if (t < o) red[t] += red[t + o];
        __syncthreads();
    }
    const float mean = red[0] * (1.0f / NC);
    __syncthreads();

    // center, store tiled f16, accumulate per-level sum of squares
    float lsum[4] = {0.f, 0.f, 0.f, 0.f};
    for (int c = t; c < NC; c += 256) {
        float v = col[(size_t)c * FS2] - mean;
        dstc[(((size_t)img * NTILE + (c >> 5)) * 32 + s) * 32 + (c & 31)] = (half_t)v;
        int l = (c < 256) ? 0 : (c < 768) ? 1 : (c < 1792) ? 2 : 3;
        lsum[l] += v * v;
    }
    #pragma unroll
    for (int l = 0; l < 4; ++l) redl[l][t] = lsum[l];
    __syncthreads();
    for (int o = 128; o > 0; o >>= 1) {
        if (t < o) {
            #pragma unroll
            for (int l = 0; l < 4; ++l) redl[l][t] += redl[l][t + o];
        }
        __syncthreads();
    }
    if (t == 0) {
        #pragma unroll
        for (int l = 0; l < 4; ++l)
            norms[(img * NLEV + l) * FS2 + s] = sqrtf(redl[l][0] + 1e-6f);
    }
}

// ---------------------------------------------------------------------------
// Kernel 2: WMMA cross-correlation + dual attention maps.
// One block (4 waves) per (q, w); wave `l` owns hyperpixel level l.
// Fragments loaded straight from the tiled layout as b128 segments:
//   A (16x32, M=pos): lane(g,r) row r holds K=[8g..8g+7] and [16+8g..+7]
//   B (32x16, N=pos): lane(g,r) col r holds K=[8g..8g+7] and [16+8g..+7]
// L2 scale applied post-accumulation: corr /= Ns[s]*Nt[t].
// ---------------------------------------------------------------------------
__global__ __launch_bounds__(128)
void corr_attn_kernel(const half_t* __restrict__ sptc,
                      const half_t* __restrict__ qryc,
                      const float*  __restrict__ snorm,
                      const float*  __restrict__ qnorm,
                      float* __restrict__ attnS, float* __restrict__ attnQ)
{
    __shared__ float Cr[4][FS2 * FS2];  // corr, per wave
    __shared__ float Pm[4][FS2 * FS2];  // softmax scratch, per wave
    __shared__ float nsh[4][FS2];
    __shared__ float nqh[4][FS2];

    const int b    = blockIdx.x;
    const int q    = b / NWAY;
    const int w    = b % NWAY;
    const int wave = threadIdx.x >> 5;
    const int lane = threadIdx.x & 31;
    const int lev  = wave;
    const int kc0  = 8 * ((1 << lev) - 1);   // first channel-block: 0,8,24,56
    const int nt   = 8 << lev;               // #channel-blocks: 8,16,32,64

    if (lane < FS2) {
        nsh[lev][lane] = snorm[(w * NLEV + lev) * FS2 + lane];
        nqh[lev][lane] = qnorm[(q * NLEV + lev) * FS2 + lane];
    }

    const int g = lane >> 4;    // half-wave group
    const int r = lane & 15;

    v8f acc00 = {0.f,0.f,0.f,0.f,0.f,0.f,0.f,0.f};
    v8f acc01 = acc00, acc10 = acc00, acc11 = acc00;

    const v8h* TA = (const v8h*)sptc + ((size_t)w * NTILE + kc0) * TILE_V;
    const v8h* TB = (const v8h*)qryc + ((size_t)q * NTILE + kc0) * TILE_V;

    // Per-lane segment indices within a tile (tile row = 4 v8h segments).
    const int aLo0 = r * 4 + g,            aHi0 = r * 4 + 2 + g;
    const int aLo1 = (16 + r) * 4 + g,     aHi1 = (16 + r) * 4 + 2 + g;
    const int bLo0 = r * 4 + g,            bHi0 = r * 4 + 2 + g;
    const int bLo1 = (16 + r) * 4 + g,     bHi1 = (16 + r) * 4 + 2 + g;

    for (int i = 0; i < nt; ++i) {
        const v8h* tA = TA + (size_t)i * TILE_V;
        const v8h* tB = TB + (size_t)i * TILE_V;
        v8h al0 = tA[aLo0], ah0 = tA[aHi0];
        v8h al1 = tA[aLo1], ah1 = tA[aHi1];
        v8h bl0 = tB[bLo0], bh0 = tB[bHi0];
        v8h bl1 = tB[bLo1], bh1 = tB[bHi1];
        v16h a0 = __builtin_shufflevector(al0, ah0, 0,1,2,3,4,5,6,7,8,9,10,11,12,13,14,15);
        v16h a1 = __builtin_shufflevector(al1, ah1, 0,1,2,3,4,5,6,7,8,9,10,11,12,13,14,15);
        v16h b0 = __builtin_shufflevector(bl0, bh0, 0,1,2,3,4,5,6,7,8,9,10,11,12,13,14,15);
        v16h b1 = __builtin_shufflevector(bl1, bh1, 0,1,2,3,4,5,6,7,8,9,10,11,12,13,14,15);
        acc00 = __builtin_amdgcn_wmma_f32_16x16x32_f16(false, a0, false, b0, (short)0, acc00, false, false);
        acc01 = __builtin_amdgcn_wmma_f32_16x16x32_f16(false, a0, false, b1, (short)0, acc01, false, false);
        acc10 = __builtin_amdgcn_wmma_f32_16x16x32_f16(false, a1, false, b0, (short)0, acc10, false, false);
        acc11 = __builtin_amdgcn_wmma_f32_16x16x32_f16(false, a1, false, b1, (short)0, acc11, false, false);
    }

    // Scatter accumulators to LDS with the L2 normalization applied.
    // C/D layout: lane L holds N = L&15 (+16*tile), VGPR v -> M = v + 8*(L>>4).
    #pragma unroll
    for (int v = 0; v < 8; ++v) {
        const int s0 = v + 8 * g;
        const int s1 = 16 + s0;
        const int t0 = r;
        const int t1 = 16 + r;
        Cr[lev][s0 * FS2 + t0] = acc00[v] / (nsh[lev][s0] * nqh[lev][t0]);
        if (t1 < FS2)
            Cr[lev][s0 * FS2 + t1] = acc01[v] / (nsh[lev][s0] * nqh[lev][t1]);
        if (s1 < FS2)
            Cr[lev][s1 * FS2 + t0] = acc10[v] / (nsh[lev][s1] * nqh[lev][t0]);
        if (s1 < FS2 && t1 < FS2)
            Cr[lev][s1 * FS2 + t1] = acc11[v] / (nsh[lev][s1] * nqh[lev][t1]);
    }

    const float invT = 1.0f / 5.0f;   // 1 / temperature_attn

    // ---- attn_s: normalize+softmax over s (per column t), then sum over t.
    if (lane < FS2) {
        const int t = lane;
        float mean = 0.f, sq = 0.f;
        for (int s = 0; s < FS2; ++s) {
            float x = Cr[lev][s * FS2 + t];
            mean += x; sq += x * x;
        }
        mean *= (1.0f / FS2);
        float var = (sq - FS2 * mean * mean) * (1.0f / (FS2 - 1));   // ddof=1
        float inv = rsqrtf(var + 1e-5f) * invT;
        float mx = -1e30f;
        for (int s = 0; s < FS2; ++s) {
            float z = (Cr[lev][s * FS2 + t] - mean) * inv;
            mx = fmaxf(mx, z);
        }
        float sum = 0.f;
        for (int s = 0; s < FS2; ++s) {
            float e = expf((Cr[lev][s * FS2 + t] - mean) * inv - mx);
            Pm[lev][s * FS2 + t] = e;
            sum += e;
        }
        float rs = 1.0f / sum;
        for (int s = 0; s < FS2; ++s) Pm[lev][s * FS2 + t] *= rs;
    }
    if (lane < FS2) {
        const int s = lane;
        float a = 0.f;
        for (int t = 0; t < FS2; ++t) a += Pm[lev][s * FS2 + t];
        attnS[(((size_t)q * NWAY + w) * NLEV + lev) * FS2 + s] = a;
    }

    // ---- attn_q: normalize+softmax over t (per row s), then sum over s.
    if (lane < FS2) {
        const int s = lane;
        float mean = 0.f, sq = 0.f;
        for (int t = 0; t < FS2; ++t) {
            float x = Cr[lev][s * FS2 + t];
            mean += x; sq += x * x;
        }
        mean *= (1.0f / FS2);
        float var = (sq - FS2 * mean * mean) * (1.0f / (FS2 - 1));
        float inv = rsqrtf(var + 1e-5f) * invT;
        float mx = -1e30f;
        for (int t = 0; t < FS2; ++t) {
            float z = (Cr[lev][s * FS2 + t] - mean) * inv;
            mx = fmaxf(mx, z);
        }
        float sum = 0.f;
        for (int t = 0; t < FS2; ++t) {
            float e = expf((Cr[lev][s * FS2 + t] - mean) * inv - mx);
            Pm[lev][s * FS2 + t] = e;
            sum += e;
        }
        float rs = 1.0f / sum;
        for (int t = 0; t < FS2; ++t) Pm[lev][s * FS2 + t] *= rs;
    }
    if (lane < FS2) {
        const int t = lane;
        float a = 0.f;
        for (int s = 0; s < FS2; ++s) a += Pm[lev][s * FS2 + t];
        attnQ[(((size_t)q * NWAY + w) * NLEV + lev) * FS2 + t] = a;
    }
}

// ---------------------------------------------------------------------------
// Kernel 3: attention-weighted pooling + cosine similarity / 0.2.
// One block per (q, w). Tiled layout: element (img,c,pos) at
// ((img*NTILE + c/32)*32 + pos)*32 + (c%32).
// ---------------------------------------------------------------------------
__global__ __launch_bounds__(256)
void pool_kernel(const half_t* __restrict__ sptc, const half_t* __restrict__ qryc,
                 const float* __restrict__ attnS, const float* __restrict__ attnQ,
                 float* __restrict__ out)
{
    __shared__ float aS[NLEV * FS2];
    __shared__ float aQ[NLEV * FS2];
    __shared__ float rdot[256], rss[256], rqq[256];

    const int b = blockIdx.x;
    const int q = b / NWAY;
    const int w = b % NWAY;
    const int t = threadIdx.x;

    if (t < NLEV * FS2) {
        aS[t] = attnS[((size_t)q * NWAY + w) * NLEV * FS2 + t];
        aQ[t] = attnQ[((size_t)q * NWAY + w) * NLEV * FS2 + t];
    }
    __syncthreads();

    float dot = 0.f, ss = 0.f, qq = 0.f;
    for (int c = t; c < NC; c += 256) {
        const int l  = (c < 256) ? 0 : (c < 768) ? 1 : (c < 1792) ? 2 : 3;
        const int kc = c >> 5, kk = c & 31;
        const half_t* sp = sptc + (((size_t)w * NTILE + kc) * 32) * 32 + kk;
        const half_t* qp = qryc + (((size_t)q * NTILE + kc) * 32) * 32 + kk;
        float sv = 0.f, qv = 0.f;
        #pragma unroll
        for (int i = 0; i < FS2; ++i) {
            sv += aS[l * FS2 + i] * (float)sp[i * 32];
            qv += aQ[l * FS2 + i] * (float)qp[i * 32];
        }
        sv *= (1.0f / FS2);
        qv *= (1.0f / FS2);
        dot += sv * qv;
        ss  += sv * sv;
        qq  += qv * qv;
    }
    rdot[t] = dot; rss[t] = ss; rqq[t] = qq;
    __syncthreads();
    for (int o = 128; o > 0; o >>= 1) {
        if (t < o) { rdot[t] += rdot[t+o]; rss[t] += rss[t+o]; rqq[t] += rqq[t+o]; }
        __syncthreads();
    }
    if (t == 0) {
        float ns  = fmaxf(sqrtf(rss[0]), 1e-8f);
        float nqn = fmaxf(sqrtf(rqq[0]), 1e-8f);
        out[q * NWAY + w] = rdot[0] / (ns * nqn) * (1.0f / 0.2f);
    }
}

// ---------------------------------------------------------------------------
extern "C" void kernel_launch(void* const* d_in, const int* in_sizes, int n_in,
                              void* d_out, int out_size, void* d_ws, size_t ws_size,
                              hipStream_t stream) {
    const float* spt = (const float*)d_in[0];   // [1,5,3840,5,5]
    const float* qry = (const float*)d_in[1];   // [75,3840,5,5]
    float* out = (float*)d_out;                 // [75,5]

    // Workspace carve-up (256B aligned; tiled f16 buffers are 2KB-tile aligned)
    auto align256 = [](size_t x) { return (x + 255) & ~(size_t)255; };
    char* ws = (char*)d_ws;
    size_t o = 0;
    const size_t sptc_bytes = (size_t)NWAY * NTILE * TILE_E * sizeof(half_t);
    const size_t qryc_bytes = (size_t)NQ   * NTILE * TILE_E * sizeof(half_t);
    half_t* sptc = (half_t*)(ws + o); o = align256(o + sptc_bytes);
    half_t* qryc = (half_t*)(ws + o); o = align256(o + qryc_bytes);
    float*  snrm = (float*) (ws + o); o = align256(o + (size_t)NWAY * NLEV * FS2 * sizeof(float));
    float*  qnrm = (float*) (ws + o); o = align256(o + (size_t)NQ   * NLEV * FS2 * sizeof(float));
    float*  atS  = (float*) (ws + o); o = align256(o + (size_t)NQ * NWAY * NLEV * FS2 * sizeof(float));
    float*  atQ  = (float*) (ws + o); o = align256(o + (size_t)NQ * NWAY * NLEV * FS2 * sizeof(float));
    (void)ws_size; (void)in_sizes; (void)n_in; (void)out_size;

    // Zero the tiled buffers so pad rows (pos 25..31) contribute nothing.
    hipMemsetAsync(sptc, 0, sptc_bytes, stream);
    hipMemsetAsync(qryc, 0, qryc_bytes, stream);

    prep_kernel<<<NWAY * FS2, 256, 0, stream>>>(spt, sptc, snrm);
    prep_kernel<<<NQ   * FS2, 256, 0, stream>>>(qry, qryc, qnrm);
    corr_attn_kernel<<<NQ * NWAY, 128, 0, stream>>>(sptc, qryc, snrm, qnrm, atS, atQ);
    pool_kernel<<<NQ * NWAY, 256, 0, stream>>>(sptc, qryc, atS, atQ, out);
}
